// BaseMoERouter_87308095193744
// MI455X (gfx1250) — matrix operand order repaired and checked by reading, not compiled, not run on hardware
//
#include <hip/hip_runtime.h>
#include <hip/hip_bf16.h>

// ---------------------------------------------------------------------------
// MoE router for MI455X (gfx1250, wave32, WMMA).
//   B=2, S=2048, H=4096, E=8, K=2, capacity = 1536
// Outputs (concatenated float32): dispatch[4096,8,1536], combine[4096,8,1536],
//                                 router_probs[4096,8], aux_loss[1]
// ---------------------------------------------------------------------------

#define H_DIM 4096
#define NTOK  4096          // B*S
#define NEXP  8
#define TOPK  2
#define CAP   1536
#define NSEL  (NTOK * TOPK) // 8192

typedef __bf16 v16bf __attribute__((ext_vector_type(16)));
typedef float  v8f   __attribute__((ext_vector_type(8)));

// ---------------------------------------------------------------------------
// Kernel 0: zero the dispatch+combine region (float4 stores -> b128)
// ---------------------------------------------------------------------------
__global__ void zero_kernel(float4* __restrict__ p, long n4) {
    long i      = (long)blockIdx.x * blockDim.x + threadIdx.x;
    long stride = (long)gridDim.x * blockDim.x;
    float4 z = make_float4(0.f, 0.f, 0.f, 0.f);
    for (; i < n4; i += stride) p[i] = z;
}

// ---------------------------------------------------------------------------
// Kernel 1: h = relu(x @ W1 + b1)  via v_wmma_f32_16x16x32_bf16
//   Block tile 128(M) x 128(N), BK=32.  256 threads = 8 waves (4x2),
//   each wave computes a 32x64 tile = 2x4 WMMA accumulators (8 WMMA/K-step).
//   Double-buffered LDS; fragment-friendly layouts so every fragment load
//   is a 16B-aligned ds_load_b128 (no VALU repacking).
// ---------------------------------------------------------------------------
#define BM 128
#define BN 128
#define BK 32
#define BK_PAD 40   // 80 B rows: 16B-aligned, bank-spread

union V16 { uint4 q[2]; v16bf v; };
union PK4 { __bf16 h[4]; uint2 u; };
union PK8 { __bf16 h[8]; uint4 u; };

__global__ __launch_bounds__(256) void gemm1_relu(const float* __restrict__ X,
                                                  const float* __restrict__ W1,
                                                  const float* __restrict__ b1,
                                                  float* __restrict__ Hout) {
    // A: row-major [m][k]; B: transposed [n][k]  (both bf16, padded rows)
    __shared__ __bf16 As[2][BM][BK_PAD];   // 2*128*40*2 = 20 KB
    __shared__ __bf16 Bt[2][BN][BK_PAD];   // 2*128*40*2 = 20 KB

    const int tid  = threadIdx.x;
    const int lane = tid & 31;
    const int wave = tid >> 5;
    const int wm   = wave & 3;   // 4 waves along M (32 rows each)
    const int wn   = wave >> 2;  // 2 waves along N (64 cols each)
    const int m0   = blockIdx.y * BM;
    const int n0   = blockIdx.x * BN;

    // --- per-thread staging coordinates (constant across K-steps) ---
    const int ar0 = tid >> 3;           // A rows: ar0, +32, +64, +96
    const int ac4 = (tid & 7) * 4;      // A col group (4 f32)
    const int bn_ = tid & 127;          // B: n index
    const int bk0 = (tid >> 7) * 16;    // B: 16 consecutive k (0 or 16)

    float4 areg[4];
    float  breg[16];

    auto load_global = [&](int kt) {
        #pragma unroll
        for (int j = 0; j < 4; ++j)
            areg[j] = *reinterpret_cast<const float4*>(
                X + (long)(m0 + ar0 + 32 * j) * H_DIM + kt + ac4);
        #pragma unroll
        for (int j = 0; j < 16; ++j)
            breg[j] = W1[(long)(kt + bk0 + j) * H_DIM + n0 + bn_];
    };
    auto store_lds = [&](int buf) {
        #pragma unroll
        for (int j = 0; j < 4; ++j) {
            PK4 pk;
            pk.h[0] = (__bf16)areg[j].x;
            pk.h[1] = (__bf16)areg[j].y;
            pk.h[2] = (__bf16)areg[j].z;
            pk.h[3] = (__bf16)areg[j].w;
            *reinterpret_cast<uint2*>(&As[buf][ar0 + 32 * j][ac4]) = pk.u;
        }
        #pragma unroll
        for (int g = 0; g < 2; ++g) {
            PK8 pk;
            #pragma unroll
            for (int j = 0; j < 8; ++j) pk.h[j] = (__bf16)breg[g * 8 + j];
            *reinterpret_cast<uint4*>(&Bt[buf][bn_][bk0 + g * 8]) = pk.u;
        }
    };

    v8f c[2][4];
    #pragma unroll
    for (int i = 0; i < 2; ++i)
        #pragma unroll
        for (int j = 0; j < 4; ++j)
            c[i][j] = (v8f){0.f, 0.f, 0.f, 0.f, 0.f, 0.f, 0.f, 0.f};

    const int mrow = lane & 15;    // A fragment row / B fragment col / C col
    const int kh   = lane >> 4;    // K half selector

    // --- prologue: stage tile 0 ---
    load_global(0);
    store_lds(0);
    __syncthreads();

    int buf = 0;
    for (int kt = 0; kt < H_DIM; kt += BK) {
        const bool has_next = (kt + BK) < H_DIM;
        if (has_next) load_global(kt + BK);   // overlap with fragment reads

        // --- fragment loads: pure ds_load_b128, layouts match ISA exactly ---
        V16 a[2], b[4];
        #pragma unroll
        for (int t = 0; t < 2; ++t) {
            const __bf16* arow = &As[buf][wm * 32 + t * 16 + mrow][0];
            a[t].q[0] = *reinterpret_cast<const uint4*>(arow + kh * 8);       // K 0..7 / 8..15
            a[t].q[1] = *reinterpret_cast<const uint4*>(arow + 16 + kh * 8);  // K 16..23 / 24..31
        }
        #pragma unroll
        for (int t = 0; t < 4; ++t) {
            const __bf16* brow = &Bt[buf][wn * 64 + t * 16 + mrow][0];
            b[t].q[0] = *reinterpret_cast<const uint4*>(brow + kh * 8);       // K 0..7 / 8..15
            b[t].q[1] = *reinterpret_cast<const uint4*>(brow + 16 + kh * 8);
        }

        if (has_next) store_lds(buf ^ 1);     // other buffer: no read conflict

        #pragma unroll
        for (int tm = 0; tm < 2; ++tm)
            #pragma unroll
            for (int tn = 0; tn < 4; ++tn)
                c[tm][tn] = __builtin_amdgcn_wmma_f32_16x16x32_bf16(
                    /*neg_a=*/false, a[tm].v, /*neg_b=*/false, b[tn].v,
                    /*c_mod=*/(short)0, c[tm][tn],
                    /*reuse_a=*/false, /*reuse_b=*/false);
        __syncthreads();
        buf ^= 1;
    }

    // --- epilogue: bias + ReLU, store f32 (C layout: lane = N, vgpr = M) ---
    const int mh = lane >> 4;
    #pragma unroll
    for (int tm = 0; tm < 2; ++tm)
        #pragma unroll
        for (int tn = 0; tn < 4; ++tn) {
            const int gcol = n0 + wn * 64 + tn * 16 + mrow;
            const float bias = b1[gcol];
            #pragma unroll
            for (int i = 0; i < 8; ++i) {
                const int grow = m0 + wm * 32 + tm * 16 + mh * 8 + i;
                float v = c[tm][tn][i] + bias;
                Hout[(long)grow * H_DIM + gcol] = v > 0.f ? v : 0.f;
            }
        }
}

// ---------------------------------------------------------------------------
// Kernel 2: logits = h @ W2 + b2, softmax, top-2 (normalized).
//   One wave per token; cross-lane reduce with wave32 shuffles.
// ---------------------------------------------------------------------------
__global__ __launch_bounds__(256) void router_kernel(const float* __restrict__ Hbuf,
                                                     const float* __restrict__ W2,
                                                     const float* __restrict__ b2,
                                                     float* __restrict__ probs_out,
                                                     float* __restrict__ flat_prob,
                                                     int* __restrict__ flat_idx) {
    const int t    = blockIdx.x * 8 + (threadIdx.x >> 5);
    const int lane = threadIdx.x & 31;
    if (t >= NTOK) return;

    float acc[NEXP] = {};
    const float* hrow = Hbuf + (long)t * H_DIM;
    for (int f = lane; f < H_DIM; f += 32) {
        const float hv = hrow[f];
        const float* wr = W2 + (long)f * NEXP;
        #pragma unroll
        for (int e = 0; e < NEXP; ++e) acc[e] = fmaf(hv, wr[e], acc[e]);
    }
    #pragma unroll
    for (int e = 0; e < NEXP; ++e)
        #pragma unroll
        for (int off = 16; off > 0; off >>= 1)
            acc[e] += __shfl_xor(acc[e], off, 32);

    if (lane == 0) {
        float p[NEXP];
        float mx = -1e30f;
        #pragma unroll
        for (int e = 0; e < NEXP; ++e) {
            p[e] = acc[e] + b2[e];
            mx = p[e] > mx ? p[e] : mx;
        }
        float s = 0.f;
        #pragma unroll
        for (int e = 0; e < NEXP; ++e) { p[e] = __expf(p[e] - mx); s += p[e]; }
        const float inv = 1.f / s;
        #pragma unroll
        for (int e = 0; e < NEXP; ++e) {
            p[e] *= inv;
            probs_out[t * NEXP + e] = p[e];
        }
        // top-2 (ties -> lower index, matching jax.lax.top_k)
        int e0 = 0;
        #pragma unroll
        for (int e = 1; e < NEXP; ++e) if (p[e] > p[e0]) e0 = e;
        int e1 = (e0 == 0) ? 1 : 0;
        #pragma unroll
        for (int e = 0; e < NEXP; ++e)
            if (e != e0 && p[e] > p[e1]) e1 = e;
        const float denom = 1.f / (p[e0] + p[e1] + 1e-8f);
        flat_idx[t * 2 + 0]  = e0;
        flat_prob[t * 2 + 0] = p[e0] * denom;
        flat_idx[t * 2 + 1]  = e1;
        flat_prob[t * 2 + 1] = p[e1] * denom;
    }
}

// ---------------------------------------------------------------------------
// Kernel 3: capacity scan (order-exact sequential semantics) + scatter + aux.
// ---------------------------------------------------------------------------
__global__ __launch_bounds__(256) void dispatch_kernel(const float* __restrict__ probs,
                                                       const float* __restrict__ flat_prob,
                                                       const int* __restrict__ flat_idx,
                                                       float* __restrict__ dispatch,
                                                       float* __restrict__ combine,
                                                       float* __restrict__ aux_out) {
    __shared__ float sums[NEXP];
    const int tid = threadIdx.x;
    if (tid < NEXP) sums[tid] = 0.f;
    __syncthreads();

    float local[NEXP] = {};
    for (int t = tid; t < NTOK; t += 256) {
        #pragma unroll
        for (int e = 0; e < NEXP; ++e) local[e] += probs[t * NEXP + e];
    }
    #pragma unroll
    for (int e = 0; e < NEXP; ++e) atomicAdd(&sums[e], local[e]);
    __syncthreads();

    if (tid == 0) {
        int counts[NEXP] = {};
        for (int n = 0; n < NSEL; ++n) {
            const int e   = flat_idx[n];
            const int pos = counts[e]++;
            if (pos < CAP) {
                const long base = (long)(n >> 1) * (NEXP * CAP) + (long)e * CAP + pos;
                dispatch[base] = 1.0f;
                combine[base]  = flat_prob[n];
            }
        }
        float aux = 0.f;
        #pragma unroll
        for (int e = 0; e < NEXP; ++e)
            aux += (sums[e] / (float)NTOK) * ((float)counts[e] / (float)NSEL);
        aux_out[0] = aux * (float)NEXP;
    }
}

// ---------------------------------------------------------------------------
extern "C" void kernel_launch(void* const* d_in, const int* in_sizes, int n_in,
                              void* d_out, int out_size, void* d_ws, size_t ws_size,
                              hipStream_t stream) {
    const float* x  = (const float*)d_in[0];   // [2,2048,4096]
    const float* W1 = (const float*)d_in[1];   // [4096,4096]
    const float* b1 = (const float*)d_in[2];   // [4096]
    const float* W2 = (const float*)d_in[3];   // [4096,8]
    const float* b2 = (const float*)d_in[4];   // [8]

    float* out = (float*)d_out;
    const long DCSZ = (long)NTOK * NEXP * CAP;     // 50,331,648
    float* dispatch = out;
    float* combine  = out + DCSZ;
    float* probs    = out + 2 * DCSZ;
    float* aux      = probs + (long)NTOK * NEXP;

    // workspace: h [4096x4096 f32] + flat_prob [8192] + flat_idx [8192]
    float* h         = (float*)d_ws;
    float* flat_prob = h + (long)NTOK * H_DIM;
    int*   flat_idx  = (int*)(flat_prob + NSEL);

    // 1) zero dispatch+combine (harness poisons d_out)
    zero_kernel<<<4096, 256, 0, stream>>>((float4*)out, (2 * DCSZ) / 4);

    // 2) h = relu(x @ W1 + b1) -- WMMA bf16, 128x128 block tiles
    dim3 g1(H_DIM / BN, NTOK / BM);   // 32 x 32 blocks
    gemm1_relu<<<g1, 256, 0, stream>>>(x, W1, b1, h);

    // 3) logits/softmax/top-2
    router_kernel<<<NTOK / 8, 256, 0, stream>>>(h, W2, b2, probs, flat_prob, flat_idx);

    // 4) capacity scan + scatter + aux loss
    dispatch_kernel<<<1, 256, 0, stream>>>(probs, flat_prob, flat_idx,
                                           dispatch, combine, aux);
}
